// ICLAdaptor_4329327034648
// MI455X (gfx1250) — matrix operand order repaired and checked by reading, not compiled, not run on hardware
//
#include <hip/hip_runtime.h>

#define B_  4
#define H_  16
#define T_  4096
#define D_  64
#define NT_ 8
#define KL_ 128

// CDNA5 async global->LDS staging (ASYNCcnt path). Flip to 0 if assembler rejects.
#define USE_ASYNC_LDS 1

typedef __attribute__((ext_vector_type(16))) __bf16 v16bf;
typedef __attribute__((ext_vector_type(8)))  float  v8f;

union FragU { v16bf v; uint4 u[2]; };

__device__ __forceinline__ unsigned short f2bf(float f) {
    unsigned int u = __float_as_uint(f);
    unsigned int r = u + 0x7FFFu + ((u >> 16) & 1u);   // round-to-nearest-even
    return (unsigned short)(r >> 16);
}

// ---------------- pass 0: zero counters ----------------
__global__ void zero_counts_kernel(int* counts) {
    int i = threadIdx.x;
    if (i < B_ * NT_) counts[i] = 0;
}

// ---------------- pass 1: convert K/V to bf16 (V transposed d-major) ----------------
__global__ void convert_kv_kernel(const float* __restrict__ kv_k,
                                  const float* __restrict__ kv_v,
                                  unsigned short* __restrict__ Kb,
                                  unsigned short* __restrict__ Vt) {
    int i = blockIdx.x * blockDim.x + threadIdx.x;     // over NT_*H_*KL_*D_
    if (i >= NT_ * H_ * KL_ * D_) return;
    Kb[i] = f2bf(kv_k[i]);                             // K stays (task,h,k,d)
    int d  = i % D_;
    int k  = (i / D_) % KL_;
    int th = i / (D_ * KL_);
    Vt[((size_t)th * D_ + d) * KL_ + k] = f2bf(kv_v[i]); // V -> (task,h,d,k)
}

// ---------------- pass 2: bin tokens by task ----------------
__global__ void bin_tokens_kernel(const long long* __restrict__ trk,
                                  int* __restrict__ counts,
                                  int* __restrict__ idx) {
    int i = blockIdx.x * blockDim.x + threadIdx.x;     // over B_*T_
    if (i >= B_ * T_) return;
    int b = i / T_, t = i % T_;
    int task = (int)trk[i];
    int pos  = atomicAdd(&counts[b * NT_ + task], 1);
    idx[(b * NT_ + task) * T_ + pos] = t;
}

// ---------------- pass 3: WMMA attention; one wave owns one 16-token tile ----------------
__global__ __launch_bounds__(128) void icl_attn_kernel(
    const float* __restrict__ x, const float* __restrict__ q,
    const unsigned short* __restrict__ Kb, const unsigned short* __restrict__ Vt,
    const float* __restrict__ gates, const int* __restrict__ counts,
    const int* __restrict__ idx, float* __restrict__ out)
{
    __shared__ __align__(16) unsigned short Qs[4][16][D_];   //  8 KB  per-wave q tiles
    __shared__ __align__(16) unsigned short Ks[KL_][D_];     // 16 KB  K   (k, d)
    __shared__ __align__(16) unsigned short Vs[D_][KL_];     // 16 KB  V^T (d, k)
    __shared__ __align__(16) unsigned short Ps[4][16][KL_];  // 16 KB  per-wave probs (bf16)
    __shared__ int rowTok[4][16];

    const int tile = blockIdx.x;                 // 64-token tile
    const int b    = blockIdx.y / NT_;
    const int task = blockIdx.y % NT_;
    const int h    = blockIdx.z;

    const int cnt = counts[b * NT_ + task];
    if (tile * 64 >= cnt) return;

    const int tid  = threadIdx.x;
    const int lane = tid & 31;
    const int wave = tid >> 5;
    const int half = lane >> 4;                  // wave32 half
    const int l16  = lane & 15;
    const int row0 = tile * 64 + wave * 16;      // this wave's token rows

    if (lane < 16)
        rowTok[wave][lane] = (row0 + lane < cnt) ? idx[(b * NT_ + task) * T_ + row0 + lane] : 0;

    // ---- stage K and V^T (bf16, pre-laid-out) into LDS ----
    const size_t kvbase = (size_t)(task * H_ + h) * (KL_ * D_);
#if USE_ASYNC_LDS
    {
        unsigned kl = (unsigned)(size_t)&Ks[0][0];          // LDS byte offset (aperture low bits)
        unsigned vl = (unsigned)(size_t)&Vs[0][0];
        unsigned long long kg = (unsigned long long)(size_t)(Kb + kvbase);
        unsigned long long vg = (unsigned long long)(size_t)(Vt + kvbase);
        for (int i = tid; i < KL_ * D_ / 8; i += 128) {
            unsigned off = (unsigned)i * 16u;
            asm volatile("global_load_async_to_lds_b128 %0, %1, off"
                         :: "v"(kl + off), "v"(kg + (unsigned long long)off) : "memory");
            asm volatile("global_load_async_to_lds_b128 %0, %1, off"
                         :: "v"(vl + off), "v"(vg + (unsigned long long)off) : "memory");
        }
        asm volatile("s_wait_asynccnt 0x0" ::: "memory");
    }
#else
    {
        const uint4* gk = (const uint4*)(Kb + kvbase);
        const uint4* gv = (const uint4*)(Vt + kvbase);
        uint4* lk = (uint4*)&Ks[0][0];
        uint4* lv = (uint4*)&Vs[0][0];
        for (int i = tid; i < KL_ * D_ / 8; i += 128) { lk[i] = gk[i]; lv[i] = gv[i]; }
    }
#endif
    __syncthreads();

    if (row0 >= cnt) return;                     // wave has no tokens (uniform per wave)
    const int nrows = min(16, cnt - row0);

    // ---- gather this wave's 16 q rows, fold 1/sqrt(D)=0.125, convert bf16 ----
    for (int e = lane; e < 16 * 16; e += 32) {   // 16 rows x 16 float4
        int r = e >> 4, c4 = (e & 15) << 2;
        float4 v4 = make_float4(0.f, 0.f, 0.f, 0.f);
        if (r < nrows) {
            const float4* src = (const float4*)(q + (((size_t)b * H_ + h) * T_ + rowTok[wave][r]) * D_);
            v4 = src[c4 >> 2];
        }
        unsigned short* dst = &Qs[wave][r][c4];
        dst[0] = f2bf(v4.x * 0.125f); dst[1] = f2bf(v4.y * 0.125f);
        dst[2] = f2bf(v4.z * 0.125f); dst[3] = f2bf(v4.w * 0.125f);
    }

    // ---- GEMM1: S[16x128] = Q x K^T, scores held in registers (8 x v8f) ----
    const unsigned short* qrow = &Qs[wave][l16][0];          // A: M = l16
    FragU a0, a1;
    a0.u[0] = *(const uint4*)(qrow + half * 8);              // K = half*8 .. +7
    a0.u[1] = *(const uint4*)(qrow + 16 + half * 8);         // K = 16+half*8 .. +7
    a1.u[0] = *(const uint4*)(qrow + 32 + half * 8);
    a1.u[1] = *(const uint4*)(qrow + 48 + half * 8);

    v8f sc[8];
    #pragma unroll
    for (int nt = 0; nt < 8; ++nt) {
        const unsigned short* brow = &Ks[nt * 16 + l16][0];  // B: N = l16 (key pos)
        FragU b0, b1;
        b0.u[0] = *(const uint4*)(brow + half * 8);
        b0.u[1] = *(const uint4*)(brow + 16 + half * 8);
        b1.u[0] = *(const uint4*)(brow + 32 + half * 8);
        b1.u[1] = *(const uint4*)(brow + 48 + half * 8);
        v8f acc = {};
        acc = __builtin_amdgcn_wmma_f32_16x16x32_bf16(false, a0.v, false, b0.v, (short)0, acc, false, false);
        acc = __builtin_amdgcn_wmma_f32_16x16x32_bf16(false, a1.v, false, b1.v, (short)0, acc, false, false);
        sc[nt] = acc;
    }

    // ---- softmax in registers: row M = r + 8*half lives in 16 lanes of one half ----
    // xor masks 1,2,4,8 only permute lane bits 0-3 -> stay within the half
    const float g = gates[task];
    #pragma unroll
    for (int r = 0; r < 8; ++r) {
        float m = sc[0][r];
        #pragma unroll
        for (int nt = 1; nt < 8; ++nt) m = fmaxf(m, sc[nt][r]);
        #pragma unroll
        for (int mask = 1; mask < 16; mask <<= 1) m = fmaxf(m, __shfl_xor(m, mask, 32));
        float s = 0.f;
        #pragma unroll
        for (int nt = 0; nt < 8; ++nt) { float e = __expf(sc[nt][r] - m); sc[nt][r] = e; s += e; }
        #pragma unroll
        for (int mask = 1; mask < 16; mask <<= 1) s += __shfl_xor(s, mask, 32);
        float w = g / s;                          // fold gate into probabilities
        #pragma unroll
        for (int nt = 0; nt < 8; ++nt)
            Ps[wave][r + half * 8][nt * 16 + l16] = f2bf(sc[nt][r] * w);
    }

    // ---- GEMM2: O[16x64] = P x V ; scatter gated residual directly from accumulators ----
    const unsigned short* prow = &Ps[wave][l16][0];          // A: M = l16
    FragU p0, p1, p2, p3;
    p0.u[0] = *(const uint4*)(prow +       half * 8); p0.u[1] = *(const uint4*)(prow +  16 + half * 8);
    p1.u[0] = *(const uint4*)(prow +  32 + half * 8); p1.u[1] = *(const uint4*)(prow +  48 + half * 8);
    p2.u[0] = *(const uint4*)(prow +  64 + half * 8); p2.u[1] = *(const uint4*)(prow +  80 + half * 8);
    p3.u[0] = *(const uint4*)(prow +  96 + half * 8); p3.u[1] = *(const uint4*)(prow + 112 + half * 8);

    #pragma unroll
    for (int dt = 0; dt < 4; ++dt) {
        const unsigned short* vrow = &Vs[dt * 16 + l16][0];  // B: N = l16 (d col), V^T rows
        v8f acc = {};
        #pragma unroll
        for (int kc = 0; kc < 4; ++kc) {
            FragU bb;
            bb.u[0] = *(const uint4*)(vrow + kc * 32 + half * 8);
            bb.u[1] = *(const uint4*)(vrow + kc * 32 + 16 + half * 8);
            const FragU& ap = (kc == 0) ? p0 : (kc == 1) ? p1 : (kc == 2) ? p2 : p3;
            acc = __builtin_amdgcn_wmma_f32_16x16x32_bf16(false, ap.v, false, bb.v, (short)0, acc, false, false);
        }
        #pragma unroll
        for (int r = 0; r < 8; ++r) {
            int row = r + half * 8;
            if (row < nrows) {
                size_t off = (((size_t)b * H_ + h) * T_ + rowTok[wave][row]) * D_ + dt * 16 + l16;
                out[off] = x[off] + acc[r];
            }
        }
    }
}

extern "C" void kernel_launch(void* const* d_in, const int* in_sizes, int n_in,
                              void* d_out, int out_size, void* d_ws, size_t ws_size,
                              hipStream_t stream) {
    (void)in_sizes; (void)n_in; (void)out_size; (void)ws_size;
    const float*     x     = (const float*)d_in[0];
    const float*     q     = (const float*)d_in[1];
    const float*     kv_k  = (const float*)d_in[2];
    const float*     kv_v  = (const float*)d_in[3];
    const float*     gates = (const float*)d_in[4];
    const long long* trk   = (const long long*)d_in[5];   // int64 task ids
    float*           out   = (float*)d_out;

    char* ws = (char*)d_ws;
    const size_t kvBytes = (size_t)NT_ * H_ * KL_ * D_ * sizeof(unsigned short); // 2 MB
    unsigned short* Kb     = (unsigned short*)ws;
    unsigned short* Vt     = (unsigned short*)(ws + kvBytes);
    int*            counts = (int*)(ws + 2 * kvBytes);
    int*            idxb   = (int*)(ws + 2 * kvBytes + 256);

    zero_counts_kernel<<<1, 32, 0, stream>>>(counts);

    const int nkv = NT_ * H_ * KL_ * D_;
    convert_kv_kernel<<<(nkv + 255) / 256, 256, 0, stream>>>(kv_k, kv_v, Kb, Vt);

    bin_tokens_kernel<<<(B_ * T_ + 255) / 256, 256, 0, stream>>>(trk, counts, idxb);

    dim3 grid(T_ / 64, B_ * NT_, H_);   // (64-token tile, b*task, h); empty tiles exit early
    icl_attn_kernel<<<grid, 128, 0, stream>>>(x, q, Kb, Vt, gates, counts, idxb, out);
}
